// DifferentialTransformer_45217415692381
// MI455X (gfx1250) — matrix operand order repaired and checked by reading, not compiled, not run on hardware
//
#include <hip/hip_runtime.h>
#include <hip/hip_bf16.h>

// ---------------------------------------------------------------------------
// Differential Transformer attention for MI455X (gfx1250, wave32, WMMA).
// B=1, N=4096, C=256, 8 heads, hd=32.  All matmuls run on
// v_wmma_f32_16x16x32_bf16 (bf16 in, f32 accumulate).  Attention is
// flash-style: one wave owns a 16-query tile of one head, streams keys in
// 64-wide blocks (4 score tiles + 4 PV WMMAs per iteration), online softmax
// in f32 with cross-lane reductions amortized over the whole 64-key block.
// ---------------------------------------------------------------------------

typedef __attribute__((ext_vector_type(16))) __bf16 v16bf;
typedef __attribute__((ext_vector_type(8)))  __bf16 v8bf;
typedef __attribute__((ext_vector_type(8)))  float  v8f;

#define N_TOK 4096
#define C_DIM 256
#define HEADS 8
#define HD    32
#define W_QKV_ROWS 768

// ---- WMMA fragment helpers (layouts per CDNA5 ISA 7.12.2) ------------------

// A-matrix 16x32 bf16 from row-major [row][ld] storage.
// lane<16: row=lane, K = {0..7, 16..23};  lane>=16: row=lane-16, K = {8..15, 24..31}
__device__ __forceinline__ v16bf load_A16x32(const __bf16* p, int ld, int row0,
                                             int col0, int lane) {
  int half = lane >> 4, r = lane & 15;
  const __bf16* b0 = p + (size_t)(row0 + r) * ld + col0 + half * 8;
  v8bf lo = *(const v8bf*)(b0);
  v8bf hi = *(const v8bf*)(b0 + 16);
  return __builtin_shufflevector(lo, hi, 0, 1, 2, 3, 4, 5, 6, 7,
                                 8, 9, 10, 11, 12, 13, 14, 15);
}

// B-matrix 32x16 bf16.  p is stored so p[nIdx*ld + kIdx] == B[K=kIdx][N=nIdx].
// lane<16: N=lane, K = {0..7, 16..23};  lane>=16: N=lane-16, K = {8..15, 24..31}.
__device__ __forceinline__ v16bf load_B32x16(const __bf16* p, int ld, int ncol0,
                                             int k0, int lane) {
  int half = lane >> 4, c = lane & 15;
  const __bf16* b0 = p + (size_t)(ncol0 + c) * ld + k0 + half * 8;
  v8bf lo = *(const v8bf*)(b0);
  v8bf hi = *(const v8bf*)(b0 + 16);
  return __builtin_shufflevector(lo, hi, 0, 1, 2, 3, 4, 5, 6, 7,
                                 8, 9, 10, 11, 12, 13, 14, 15);
}

__device__ __forceinline__ v8f wmma_bf16(v16bf a, v16bf b, v8f c) {
  return __builtin_amdgcn_wmma_f32_16x16x32_bf16(false, a, false, b,
                                                 (short)0, c, false, false);
}

// ---- Kernel 1: differentials + bf16 conversion of activations --------------
// f2d[c, n] = x_flat[c*4096 + n]  (the reference's reshape, NOT a permute)
__global__ void __launch_bounds__(256)
prep_kernel(const float* __restrict__ x, __bf16* __restrict__ xb,
            __bf16* __restrict__ hb, __bf16* __restrict__ vb) {
  int idx = blockIdx.x * 256 + threadIdx.x;     // over N*C
  if (idx >= N_TOK * C_DIM) return;
  int n = idx >> 8, c = idx & 255;
  xb[idx] = (__bf16)x[idx];
  size_t base = (size_t)c * N_TOK + n;          // flat index into (C,64,64) view
  float f0 = x[base];
  float h = ((n & 63) == 63) ? 0.0f : x[base + 1] - f0;    // d/dx, last col -> 0
  float v = ((n >> 6) == 63) ? 0.0f : x[base + 64] - f0;   // d/dy, last row -> 0
  hb[idx] = (__bf16)h;
  vb[idx] = (__bf16)v;
}

// ---- Kernel 2: weight conversion to bf16 -----------------------------------
__global__ void __launch_bounds__(256)
wconv_kernel(const float* __restrict__ wq, const float* __restrict__ wp,
             __bf16* __restrict__ wqb, __bf16* __restrict__ wpb) {
  int idx = blockIdx.x * 256 + threadIdx.x;
  if (idx < W_QKV_ROWS * C_DIM) wqb[idx] = (__bf16)wq[idx];
  if (idx < C_DIM * C_DIM)      wpb[idx] = (__bf16)wp[idx];
}

// ---- Kernel 3: QKV projections (7 logical GEMMs, all WMMA) -----------------
// Qall/Kall: [variant(3)][head][n][hd] bf16.   Vt: [head][hd][n] bf16
// (V stored transposed so the PV B-fragment is two contiguous 16B loads/lane).
__global__ void __launch_bounds__(128)
qkv_kernel(const __bf16* __restrict__ xb, const __bf16* __restrict__ hb,
           const __bf16* __restrict__ vb, const __bf16* __restrict__ wqb,
           __bf16* __restrict__ Qall, __bf16* __restrict__ Kall,
           __bf16* __restrict__ Vt) {
  int wid = threadIdx.x >> 5, lane = threadIdx.x & 31;
  int task = blockIdx.x * 4 + wid;              // 112 out-tiles * 256 n-tiles
  if (task >= 112 * 256) return;
  int n0 = (task & 255) * 16;
  int ot = task >> 8;                           // 0..111
  int t, out0;
  if (ot < 48)      { t = 0; out0 = ot * 16; }          // x: q,k,v (cols 0..767)
  else if (ot < 80) { t = 1; out0 = (ot - 48) * 16; }   // hdiff: q,k (0..511)
  else              { t = 2; out0 = (ot - 80) * 16; }   // vdiff: q,k (0..511)
  const __bf16* T = (t == 0) ? xb : (t == 1) ? hb : vb;

  v8f acc = {};
#pragma unroll
  for (int c0 = 0; c0 < C_DIM; c0 += 32) {
    v16bf a = load_A16x32(T, C_DIM, n0, c0, lane);
    v16bf b = load_B32x16(wqb, C_DIM, out0, c0, lane);  // B[k=c][n=out] = W[out][c]
    acc = wmma_bf16(a, b, acc);
  }

  int half = lane >> 4, nc = lane & 15;
  int out = out0 + nc;
  int sel = out >> 8;                          // 0:q 1:k 2:v
  int cc = out & 255;
  int head = cc >> 5, d = cc & 31;
#pragma unroll
  for (int r = 0; r < 8; ++r) {
    int n = n0 + r + 8 * half;                 // C-layout: M = r + 8*half
    __bf16 val = (__bf16)acc[r];
    if (sel == 0)
      Qall[(((size_t)t * HEADS + head) * N_TOK + n) * HD + d] = val;
    else if (sel == 1)
      Kall[(((size_t)t * HEADS + head) * N_TOK + n) * HD + d] = val;
    else
      Vt[((size_t)head * HD + d) * N_TOK + n] = val;
  }
}

// ---- Kernel 4: flash attention (3-term scores + positional bias) -----------
// 64 keys per iteration: 12 score WMMAs + 4 PV WMMAs; one cross-lane
// max/sum reduction per row per 64 keys.
__global__ void __launch_bounds__(128)
attn_kernel(const __bf16* __restrict__ Qall, const __bf16* __restrict__ Kall,
            const __bf16* __restrict__ Vt, __bf16* __restrict__ Ob) {
  __shared__ __bf16 plds_all[4][16 * 64];      // 2KB P-transpose slice per wave
  int wid = threadIdx.x >> 5, lane = threadIdx.x & 31;
  int task = blockIdx.x * 4 + wid;             // 8 heads * 256 q-tiles
  if (task >= HEADS * 256) return;
  int head = task >> 8;
  int q0 = (task & 255) * 16;
  int half = lane >> 4, nc = lane & 15;
  __bf16* plds = plds_all[wid];

  const size_t hoff = (size_t)head * N_TOK * HD;
  const size_t voff = (size_t)HEADS * N_TOK * HD;
  const __bf16* Qx = Qall + hoff;
  const __bf16* Qh = Qall + voff + hoff;
  const __bf16* Qv = Qall + 2 * voff + hoff;
  const __bf16* Kx = Kall + hoff;
  const __bf16* Kh = Kall + voff + hoff;
  const __bf16* Kv = Kall + 2 * voff + hoff;
  const __bf16* Vh = Vt + (size_t)head * HD * N_TOK;

  // Q fragments: full hd=32 fits one WMMA K-dim.
  v16bf qx = load_A16x32(Qx, HD, q0, 0, lane);
  v16bf qh = load_A16x32(Qh, HD, q0, 0, lane);
  v16bf qv = load_A16x32(Qv, HD, q0, 0, lane);

  v8f acc0 = {}, acc1 = {};                    // O accumulator, d 0..15 / 16..31
  float Mr[8], Lr[8];
#pragma unroll
  for (int r = 0; r < 8; ++r) { Mr[r] = -1e30f; Lr[r] = 0.0f; }
  const float inv_scale = 0.17677669529663687f;  // 1/sqrt(32)
  const float nn_base = (float)(q0 + 8 * half);

  for (int m0 = 0; m0 < N_TOK; m0 += 64) {
    // ---- Scores: four 16x16 tiles, each = sum of 3 QK^T terms -------------
    v8f s0 = {}, s1 = {}, s2 = {}, s3 = {};
    {
      v16bf b = load_B32x16(Kx, HD, m0, 0, lane);       s0 = wmma_bf16(qx, b, s0);
      b       = load_B32x16(Kh, HD, m0, 0, lane);       s0 = wmma_bf16(qh, b, s0);
      b       = load_B32x16(Kv, HD, m0, 0, lane);       s0 = wmma_bf16(qv, b, s0);
    }
    {
      v16bf b = load_B32x16(Kx, HD, m0 + 16, 0, lane);  s1 = wmma_bf16(qx, b, s1);
      b       = load_B32x16(Kh, HD, m0 + 16, 0, lane);  s1 = wmma_bf16(qh, b, s1);
      b       = load_B32x16(Kv, HD, m0 + 16, 0, lane);  s1 = wmma_bf16(qv, b, s1);
    }
    {
      v16bf b = load_B32x16(Kx, HD, m0 + 32, 0, lane);  s2 = wmma_bf16(qx, b, s2);
      b       = load_B32x16(Kh, HD, m0 + 32, 0, lane);  s2 = wmma_bf16(qh, b, s2);
      b       = load_B32x16(Kv, HD, m0 + 32, 0, lane);  s2 = wmma_bf16(qv, b, s2);
    }
    {
      v16bf b = load_B32x16(Kx, HD, m0 + 48, 0, lane);  s3 = wmma_bf16(qx, b, s3);
      b       = load_B32x16(Kh, HD, m0 + 48, 0, lane);  s3 = wmma_bf16(qh, b, s3);
      b       = load_B32x16(Kv, HD, m0 + 48, 0, lane);  s3 = wmma_bf16(qv, b, s3);
    }

    // ---- Scale + positional bias G = 2 - |n - m|, online softmax ----------
    float mcol = (float)(m0 + nc);
#pragma unroll
    for (int r = 0; r < 8; ++r) {
      float nn = nn_base + (float)r;
      s0[r] = s0[r] * inv_scale + (2.0f - fabsf(nn - mcol));
      s1[r] = s1[r] * inv_scale + (2.0f - fabsf(nn - (mcol + 16.0f)));
      s2[r] = s2[r] * inv_scale + (2.0f - fabsf(nn - (mcol + 32.0f)));
      s3[r] = s3[r] * inv_scale + (2.0f - fabsf(nn - (mcol + 48.0f)));
    }
#pragma unroll
    for (int r = 0; r < 8; ++r) {
      float t = fmaxf(fmaxf(s0[r], s1[r]), fmaxf(s2[r], s3[r]));
#pragma unroll
      for (int mk = 1; mk < 16; mk <<= 1) t = fmaxf(t, __shfl_xor(t, mk, 32));
      float nm = fmaxf(Mr[r], t);
      float scl = __expf(Mr[r] - nm);
      Mr[r] = nm;
      s0[r] = __expf(s0[r] - nm);
      s1[r] = __expf(s1[r] - nm);
      s2[r] = __expf(s2[r] - nm);
      s3[r] = __expf(s3[r] - nm);
      float s = (s0[r] + s1[r]) + (s2[r] + s3[r]);
#pragma unroll
      for (int mk = 1; mk < 16; mk <<= 1) s += __shfl_xor(s, mk, 32);
      Lr[r] = Lr[r] * scl + s;
      acc0[r] *= scl;
      acc1[r] *= scl;
      // Stage P (C-layout -> row-major bf16) for the A-fragment reload.
      __bf16* prow = plds + (r + 8 * half) * 64 + nc;
      prow[0]  = (__bf16)s0[r];
      prow[16] = (__bf16)s1[r];
      prow[32] = (__bf16)s2[r];
      prow[48] = (__bf16)s3[r];
    }
    asm volatile("s_wait_dscnt 0x0" ::: "memory");  // same-wave LDS RAW

    // ---- PV: P(16x64) x V(64x32) as 2 A-fragments x 2 d-tiles -------------
    v16bf pf0 = load_A16x32(plds, 64, 0, 0, lane);   // m = m0 .. m0+31
    v16bf pf1 = load_A16x32(plds, 64, 0, 32, lane);  // m = m0+32 .. m0+63
    v16bf b00 = load_B32x16(Vh, N_TOK, 0, m0, lane);       // d 0..15,  m block 0
    v16bf b01 = load_B32x16(Vh, N_TOK, 16, m0, lane);      // d 16..31, m block 0
    v16bf b10 = load_B32x16(Vh, N_TOK, 0, m0 + 32, lane);  // d 0..15,  m block 1
    v16bf b11 = load_B32x16(Vh, N_TOK, 16, m0 + 32, lane); // d 16..31, m block 1
    acc0 = wmma_bf16(pf0, b00, acc0);
    acc1 = wmma_bf16(pf0, b01, acc1);
    acc0 = wmma_bf16(pf1, b10, acc0);
    acc1 = wmma_bf16(pf1, b11, acc1);
  }

#pragma unroll
  for (int r = 0; r < 8; ++r) {
    float inv = 1.0f / Lr[r];
    int n = q0 + r + 8 * half;
    size_t base = (size_t)n * C_DIM + head * HD;
    Ob[base + nc]      = (__bf16)(acc0[r] * inv);
    Ob[base + 16 + nc] = (__bf16)(acc1[r] * inv);
  }
}

// ---- Kernel 5: output projection + bias (WMMA) -----------------------------
__global__ void __launch_bounds__(128)
proj_kernel(const __bf16* __restrict__ Ob, const __bf16* __restrict__ wpb,
            const float* __restrict__ bias, float* __restrict__ out) {
  int wid = threadIdx.x >> 5, lane = threadIdx.x & 31;
  int task = blockIdx.x * 4 + wid;             // 16 out-tiles * 256 n-tiles
  if (task >= 16 * 256) return;
  int n0 = (task & 255) * 16;
  int co0 = (task >> 8) * 16;

  v8f acc = {};
#pragma unroll
  for (int c0 = 0; c0 < C_DIM; c0 += 32) {
    v16bf a = load_A16x32(Ob, C_DIM, n0, c0, lane);
    v16bf b = load_B32x16(wpb, C_DIM, co0, c0, lane);
    acc = wmma_bf16(a, b, acc);
  }
  int half = lane >> 4, nc = lane & 15;
  float bv = bias[co0 + nc];
#pragma unroll
  for (int r = 0; r < 8; ++r) {
    int n = n0 + r + 8 * half;
    out[(size_t)n * C_DIM + co0 + nc] = acc[r] + bv;
  }
}

// ---- Launch ----------------------------------------------------------------
extern "C" void kernel_launch(void* const* d_in, const int* in_sizes, int n_in,
                              void* d_out, int out_size, void* d_ws,
                              size_t ws_size, hipStream_t stream) {
  const float* x      = (const float*)d_in[0];   // (1,4096,256)
  const float* w_qkv  = (const float*)d_in[1];   // (768,256)
  const float* w_proj = (const float*)d_in[2];   // (256,256)
  const float* b_proj = (const float*)d_in[3];   // (256,)
  float* out = (float*)d_out;

  char* ws = (char*)d_ws;
  const size_t NC2 = (size_t)N_TOK * C_DIM * 2;          // bf16 activation plane
  __bf16* xb   = (__bf16*)(ws);                          // 2 MB
  __bf16* hb   = (__bf16*)(ws + NC2);                    // 2 MB
  __bf16* vb   = (__bf16*)(ws + 2 * NC2);                // 2 MB
  __bf16* wqb  = (__bf16*)(ws + 3 * NC2);                // 384 KB
  __bf16* wpb  = (__bf16*)(ws + 3 * NC2 + 393216);       // 128 KB
  __bf16* Qall = (__bf16*)(ws + 3 * NC2 + 524288);       // 6 MB (3 variants)
  __bf16* Kall = (__bf16*)(ws + 3 * NC2 + 524288 + 3 * NC2);       // 6 MB
  __bf16* Vt   = (__bf16*)(ws + 3 * NC2 + 524288 + 6 * NC2);       // 2 MB
  __bf16* Ob   = (__bf16*)(ws + 3 * NC2 + 524288 + 7 * NC2);       // 2 MB

  prep_kernel<<<(N_TOK * C_DIM) / 256, 256, 0, stream>>>(x, xb, hb, vb);
  wconv_kernel<<<(W_QKV_ROWS * C_DIM) / 256, 256, 0, stream>>>(w_qkv, w_proj,
                                                               wqb, wpb);
  qkv_kernel<<<(112 * 256) / 4, 128, 0, stream>>>(xb, hb, vb, wqb,
                                                  Qall, Kall, Vt);
  attn_kernel<<<(HEADS * 256) / 4, 128, 0, stream>>>(Qall, Kall, Vt, Ob);
  proj_kernel<<<(16 * 256) / 4, 128, 0, stream>>>(Ob, wpb, b_proj, out);
}
